// UltraEfficientRouter_11390253269260
// MI455X (gfx1250) — matrix operand order repaired and checked by reading, not compiled, not run on hardware
//
#include <hip/hip_runtime.h>

typedef _Float16 half16 __attribute__((ext_vector_type(16)));
typedef float    float8 __attribute__((ext_vector_type(8)));
typedef float    f32x4  __attribute__((ext_vector_type(4)));

// ---------------------------------------------------------------------------
// Kernel 1: 8x8 average pool.  x[16,256,256,256] -> pooled[16,256,32,32]
// One wave per pooled output row; lane = pooled column.  Bandwidth-bound:
// each wave streams 8 KB of contiguous input via b128 loads.  Input is a
// one-touch 1.07 GB stream (>> 192 MB L2) -> non-temporal loads so the L2
// keeps the small downstream buffers instead.  Output store stays RT.
// ---------------------------------------------------------------------------
__global__ void pool8_kernel(const float* __restrict__ x,
                             float* __restrict__ pooled) {
    int wid  = blockIdx.x * 8 + (threadIdx.x >> 5);   // pooled row id, 131072 total
    int lane = threadIdx.x & 31;
    int py = wid & 31;            // pooled y
    int bc = wid >> 5;            // b*256 + c
    const f32x4* in = (const f32x4*)(x + ((size_t)bc * 256 + (size_t)py * 8) * 256);
    float sum = 0.0f;
#pragma unroll
    for (int r = 0; r < 8; ++r) {
        f32x4 a = __builtin_nontemporal_load(in + r * 64 + 2 * lane);
        f32x4 c = __builtin_nontemporal_load(in + r * 64 + 2 * lane + 1);
        sum += a.x + a.y + a.z + a.w + c.x + c.y + c.z + c.w;
    }
    pooled[(size_t)bc * 1024 + py * 32 + lane] = sum * (1.0f / 64.0f);
}

// ---------------------------------------------------------------------------
// Kernel 2: depthwise 3x3 (SAME) + GroupNorm(8 groups) + SiLU.
// Block per (b, group): 32 channels x 32x32.  Two passes (stats, normalize),
// conv recomputed (cheap, pooled sits in L2).  Output f16, [b][sp][c] layout.
// ---------------------------------------------------------------------------
__device__ __forceinline__ float dw_conv_at(const float* __restrict__ pooled,
                                            const float* __restrict__ wdw,
                                            int b, int c, int e) {
    int pos = e & 1023;
    int y = pos >> 5, xx = pos & 31;
    const float* p = pooled + (size_t)(b * 256 + c) * 1024;
    const float* w = wdw + c * 9;
    float acc = 0.0f;
#pragma unroll
    for (int dy = 0; dy < 3; ++dy) {
        int yy = y + dy - 1;
        if (yy < 0 || yy > 31) continue;
#pragma unroll
        for (int dx = 0; dx < 3; ++dx) {
            int xc = xx + dx - 1;
            if (xc < 0 || xc > 31) continue;
            acc += p[yy * 32 + xc] * w[dy * 3 + dx];
        }
    }
    return acc;
}

__global__ void dwconv_gn1_silu_kernel(const float* __restrict__ pooled,
                                       const float* __restrict__ wdw,
                                       const float* __restrict__ g1,
                                       const float* __restrict__ b1,
                                       _Float16* __restrict__ h1) {
    int b = blockIdx.x >> 3;
    int g = blockIdx.x & 7;
    int t = threadIdx.x;
    __shared__ float rs[256], rq[256];
    __shared__ float s_mean, s_rstd;

    float sum = 0.0f, sq = 0.0f;
    for (int i = 0; i < 128; ++i) {
        int e = t + (i << 8);
        int c = g * 32 + (e >> 10);
        float v = dw_conv_at(pooled, wdw, b, c, e);
        sum += v; sq += v * v;
    }
    rs[t] = sum; rq[t] = sq;
    __syncthreads();
    for (int off = 128; off > 0; off >>= 1) {
        if (t < off) { rs[t] += rs[t + off]; rq[t] += rq[t + off]; }
        __syncthreads();
    }
    if (t == 0) {
        float m = rs[0] * (1.0f / 32768.0f);
        float var = rq[0] * (1.0f / 32768.0f) - m * m;
        s_mean = m; s_rstd = rsqrtf(var + 1e-5f);
    }
    __syncthreads();
    float mean = s_mean, rstd = s_rstd;

    for (int i = 0; i < 128; ++i) {
        int e = t + (i << 8);
        int pos = e & 1023;
        int c = g * 32 + (e >> 10);
        float v = dw_conv_at(pooled, wdw, b, c, e);
        v = (v - mean) * rstd * g1[c] + b1[c];
        float s = v / (1.0f + expf(-v));          // SiLU
        h1[((size_t)b * 1024 + pos) * 256 + c] = (_Float16)s;
    }
}

// ---------------------------------------------------------------------------
// Kernel 3: pointwise conv C=256 -> R=16 as WMMA GEMM.
// Per batch b:  D[16 x 1024] = W1[16 x 256] * H1[256 x 1024]
// v_wmma_f32_16x16x32_f16, K-loop of 8, 64 N-tiles (16 spatial cols each).
// A fragment (weights) hoisted; B fragment loads contiguous half-pairs from
// the [b][sp][c] f16 activation layout.  Output h2 in [b][sp][r] layout.
// ---------------------------------------------------------------------------
__global__ void pw1_wmma_kernel(const _Float16* __restrict__ h1,
                                const float* __restrict__ w1,
                                float* __restrict__ h2) {
    int b    = blockIdx.x;
    int lane = threadIdx.x & 31;
    int wave = threadIdx.x >> 5;   // 0..3
    int m  = lane & 15;            // A row (R index) / B column (spatial)
    int hi = lane >> 4;            // 0 or 1
    int abase = hi * 8;            // A: lanes>=16 hold K=8..15 / 24..31
    int bkoff = hi * 8;            // B: lanes>=16 hold K=8..15 / 24..31

    // Hoist all 8 A fragments (16-bit A 16x32 layout per ISA table).
    half16 A[8];
#pragma unroll
    for (int kk = 0; kk < 8; ++kk) {
        int k0 = kk * 32;
#pragma unroll
        for (int v = 0; v < 8; ++v) {
            int k = k0 + abase + ((v < 4) ? (2 * v) : (16 + 2 * (v - 4)));
            A[kk][2 * v]     = (_Float16)w1[m * 256 + k];
            A[kk][2 * v + 1] = (_Float16)w1[m * 256 + k + 1];
        }
    }

    const _Float16* hb = h1 + (size_t)b * 1024 * 256;
    for (int ti = 0; ti < 16; ++ti) {
        int tile = wave + 4 * ti;                 // 0..63
        const _Float16* col = hb + (size_t)(tile * 16 + m) * 256;
        float8 Cacc;
#pragma unroll
        for (int j = 0; j < 8; ++j) Cacc[j] = 0.0f;
#pragma unroll
        for (int kk = 0; kk < 8; ++kk) {
            int kb = kk * 32 + bkoff;
            half16 Bf;
#pragma unroll
            for (int v = 0; v < 8; ++v) {
                int kv = kb + ((v < 4) ? (2 * v) : (16 + 2 * (v - 4)));
                Bf[2 * v]     = col[kv];
                Bf[2 * v + 1] = col[kv + 1];
            }
            Cacc = __builtin_amdgcn_wmma_f32_16x16x32_f16(
                false, A[kk], false, Bf, (short)0, Cacc, false, false);
        }
        // D layout: VGPR j, lane -> M = j + 8*hi (R index), N = m (spatial)
        float* out = h2 + ((size_t)b * 1024 + tile * 16 + m) * 16;
#pragma unroll
        for (int j = 0; j < 8; ++j) out[j + 8 * hi] = Cacc[j];
    }
}

// ---------------------------------------------------------------------------
// Kernel 4: GroupNorm(4 groups over R=16) + SiLU + pointwise R->E(8) + bias +
// softmax over E + spatial mean + top-2 + renormalize.  Block per batch.
// ---------------------------------------------------------------------------
__global__ void finish_kernel(const float* __restrict__ h2,
                              const float* __restrict__ g2,
                              const float* __restrict__ b2,
                              const float* __restrict__ w2,
                              const float* __restrict__ bias2,
                              float* __restrict__ vals_out,
                              int* __restrict__ idx_out) {
    int b = blockIdx.x, t = threadIdx.x;
    __shared__ float smem[8 * 256];
    __shared__ float s_mu[4], s_rs[4];
    const float* hb = h2 + (size_t)b * 1024 * 16;

    // GN2 stats: each group = 4 channels x 1024 spatial = 4096 elems
    float gsum[4] = {0, 0, 0, 0}, gsq[4] = {0, 0, 0, 0};
    for (int i = 0; i < 4; ++i) {
        const float* row = hb + (t + (i << 8)) * 16;
#pragma unroll
        for (int r = 0; r < 16; ++r) {
            float v = row[r];
            gsum[r >> 2] += v; gsq[r >> 2] += v * v;
        }
    }
#pragma unroll
    for (int g = 0; g < 4; ++g) { smem[g * 256 + t] = gsum[g]; smem[(g + 4) * 256 + t] = gsq[g]; }
    __syncthreads();
    for (int off = 128; off > 0; off >>= 1) {
        if (t < off)
            for (int row = 0; row < 8; ++row) smem[row * 256 + t] += smem[row * 256 + t + off];
        __syncthreads();
    }
    if (t < 4) {
        float m = smem[t * 256] * (1.0f / 4096.0f);
        float var = smem[(t + 4) * 256] * (1.0f / 4096.0f) - m * m;
        s_mu[t] = m; s_rs[t] = rsqrtf(var + 1e-5f);
    }
    __syncthreads();
    float mu[4], rsd[4];
#pragma unroll
    for (int g = 0; g < 4; ++g) { mu[g] = s_mu[g]; rsd[g] = s_rs[g]; }

    // normalize + SiLU + logits + stable softmax, accumulate spatial mean
    float acc[8] = {0, 0, 0, 0, 0, 0, 0, 0};
    for (int i = 0; i < 4; ++i) {
        const float* row = hb + (t + (i << 8)) * 16;
        float s[16];
#pragma unroll
        for (int r = 0; r < 16; ++r) {
            int g = r >> 2;
            float v = (row[r] - mu[g]) * rsd[g] * g2[r] + b2[r];
            s[r] = v / (1.0f + expf(-v));
        }
        float logit[8]; float mx = -1e30f;
#pragma unroll
        for (int e = 0; e < 8; ++e) {
            float L = bias2[e];
#pragma unroll
            for (int r = 0; r < 16; ++r) L += w2[e * 16 + r] * s[r];
            logit[e] = L; mx = fmaxf(mx, L);
        }
        float ps = 0.0f, p[8];
#pragma unroll
        for (int e = 0; e < 8; ++e) { p[e] = expf(logit[e] - mx); ps += p[e]; }
        float inv = 1.0f / ps;
#pragma unroll
        for (int e = 0; e < 8; ++e) acc[e] += p[e] * inv;
    }
    __syncthreads();
#pragma unroll
    for (int e = 0; e < 8; ++e) smem[e * 256 + t] = acc[e];
    __syncthreads();
    for (int off = 128; off > 0; off >>= 1) {
        if (t < off)
            for (int row = 0; row < 8; ++row) smem[row * 256 + t] += smem[row * 256 + t + off];
        __syncthreads();
    }
    if (t == 0) {
        float pooled[8];
#pragma unroll
        for (int e = 0; e < 8; ++e) pooled[e] = smem[e * 256] * (1.0f / 1024.0f);
        int i1 = 0;
        for (int e = 1; e < 8; ++e) if (pooled[e] > pooled[i1]) i1 = e;
        int i2 = -1;
        for (int e = 0; e < 8; ++e) {
            if (e == i1) continue;
            if (i2 < 0 || pooled[e] > pooled[i2]) i2 = e;
        }
        float v1 = pooled[i1], v2 = pooled[i2];
        float den = v1 + v2 + 1e-9f;
        vals_out[b * 2]     = v1 / den;
        vals_out[b * 2 + 1] = v2 / den;
        idx_out[b * 2]      = i1;
        idx_out[b * 2 + 1]  = i2;
    }
}

// ---------------------------------------------------------------------------
extern "C" void kernel_launch(void* const* d_in, const int* in_sizes, int n_in,
                              void* d_out, int out_size, void* d_ws, size_t ws_size,
                              hipStream_t stream) {
    const float* x     = (const float*)d_in[0];
    const float* w_dw  = (const float*)d_in[1];
    const float* gn1_g = (const float*)d_in[2];
    const float* gn1_b = (const float*)d_in[3];
    const float* w_pw1 = (const float*)d_in[4];
    const float* gn2_g = (const float*)d_in[5];
    const float* gn2_b = (const float*)d_in[6];
    const float* w_pw2 = (const float*)d_in[7];
    const float* b_pw2 = (const float*)d_in[8];

    char* ws = (char*)d_ws;
    float*    pooled = (float*)ws;                         // 16 MB: [16][256][1024] f32
    _Float16* h1     = (_Float16*)(ws + (16u << 20));      //  8 MB: [16][1024][256] f16
    float*    h2     = (float*)(ws + (24u << 20));         //  1 MB: [16][1024][16]  f32

    pool8_kernel<<<16384, 256, 0, stream>>>(x, pooled);
    dwconv_gn1_silu_kernel<<<128, 256, 0, stream>>>(pooled, w_dw, gn1_g, gn1_b, h1);
    pw1_wmma_kernel<<<16, 128, 0, stream>>>(h1, w_pw1, h2);
    finish_kernel<<<16, 256, 0, stream>>>(h2, gn2_g, gn2_b, w_pw2, b_pw2,
                                          (float*)d_out, ((int*)d_out) + 32);
}
